// CNN_LSTM_GCN_80144089743550
// MI455X (gfx1250) — matrix operand order repaired and checked by reading, not compiled, not run on hardware
//
#include <hip/hip_runtime.h>
#include <hip/hip_bf16.h>

// ---------------------------------------------------------------------------
// CNN_LSTM_GCN on gfx1250 (MI455X). All GEMMs via v_wmma_f32_16x16x32_f16.
// Sizes (fixed by reference): N=20000, T=8, F=128, E=320000,
// H_GCN=32, H_CONV=64, H_LSTM=128.
// ---------------------------------------------------------------------------

typedef __attribute__((ext_vector_type(16))) _Float16 v16h;
typedef __attribute__((ext_vector_type(8)))  _Float16 v8h;
typedef __attribute__((ext_vector_type(8)))  float    v8f;
typedef _Float16 half_t;

#define NNODES 20000
#define TSTEPS 8
#define FEAT   128
#define NEDGES 320000
#define HGCN   32
#define HCONV  64
#define HLSTM  128

__device__ __forceinline__ v8f wmma_f16(v16h a, v16h b, v8f c) {
  return __builtin_amdgcn_wmma_f32_16x16x32_f16(false, a, false, b, (short)0, c,
                                                false, false);
}

// A-fragment (16x32 f16): lane holds row (lane&15); per the CDNA5 layout the
// lane's 16 halves are two contiguous K-runs of 8:
//   lanes 0-15 : K = [0..8)  and [16..24)
//   lanes 16-31: K = [8..16) and [24..32)
// `p` points at (row base + kchunk*32) halves, 16B aligned.
__device__ __forceinline__ v16h load_a_f16(const half_t* p, int lane) {
  const half_t* q = p + ((lane >> 4) << 3);
  v8h lo = *(const v8h*)(q);
  v8h hi = *(const v8h*)(q + 16);
  v16h a;
#pragma unroll
  for (int e = 0; e < 8; ++e) { a[e] = lo[e]; a[8 + e] = hi[e]; }
  return a;
}

// Same fragment but converting from fp32 source.
__device__ __forceinline__ v16h load_a_f32(const float* p, int lane) {
  const float* q = p + ((lane >> 4) << 3);
  v16h a;
#pragma unroll
  for (int e = 0; e < 8; ++e) {
    a[e]     = (half_t)q[e];
    a[8 + e] = (half_t)q[e + 16];
  }
  return a;
}

__device__ __forceinline__ float sigf(float x) {
  return 1.0f / (1.0f + __expf(-x));
}

// ---------------------------------------------------------------------------
// Prep kernels
// ---------------------------------------------------------------------------

__global__ void cl_fill_f32(float* p, float v, int n) {
  int i = blockIdx.x * blockDim.x + threadIdx.x;
  if (i < n) p[i] = v;
}
__global__ void cl_fill_f16(half_t* p, int n) {
  int i = blockIdx.x * blockDim.x + threadIdx.x;
  if (i < n) p[i] = (half_t)0.0f;
}

// Pack weight matrix (logical [K][O], element (k,o)) into WMMA B-fragment
// layout: out[((kc*NT + nt)*32 + lane)*16 + e] where within a 32-K chunk
// K = (lane>>4)*8 + (e&7) + (e>>3)*16 and column = nt*16 + (lane&15). One 32B load/lane later.
__global__ void cl_pack_b(const float* W, half_t* out, int K, int O, int kMajor) {
  int NT = O >> 4, KC = K >> 5;
  int total = KC * NT * 512;
  int idx = blockIdx.x * blockDim.x + threadIdx.x;
  if (idx >= total) return;
  int e = idx & 15;
  int lane = (idx >> 4) & 31;
  int nt = (idx >> 9) % NT;
  int kc = (idx >> 9) / NT;
  int k = kc * 32 + ((lane >> 4) << 3) + (e & 7) + ((e >> 3) << 4);
  int o = nt * 16 + (lane & 15);
  float w = kMajor ? W[k * O + o] : W[o * K + k];
  out[idx] = (half_t)w;
}

// Conv1d(32->64,k=2) weights as GEMM B: logical K index j = k*32+ic, O = oc.
__global__ void cl_pack_conv(const float* convW, half_t* out) {
  int idx = blockIdx.x * blockDim.x + threadIdx.x;  // total 2*4*512 = 4096
  if (idx >= 4096) return;
  int e = idx & 15;
  int lane = (idx >> 4) & 31;
  int nt = (idx >> 9) % 4;
  int kc = (idx >> 9) / 4;
  int j = kc * 32 + ((lane >> 4) << 3) + (e & 7) + ((e >> 3) << 4);
  int oc = nt * 16 + (lane & 15);
  int ic = j & 31, k = j >> 5;
  out[idx] = (half_t)convW[(oc * 32 + ic) * 2 + k];
}

__global__ void cl_bias_sum(const float* bi, const float* bh, float* bz) {
  int i = blockIdx.x * blockDim.x + threadIdx.x;
  if (i < 4 * HLSTM) bz[i] = bi[i] + bh[i];
}

__global__ void cl_degree(const int* ei, float* deg) {
  int e = blockIdx.x * blockDim.x + threadIdx.x;
  if (e < NEDGES) atomicAdd(&deg[ei[NEDGES + e]], 1.0f);  // col = target
}

__global__ void cl_dinv(const float* deg, float* dinv) {
  int i = blockIdx.x * blockDim.x + threadIdx.x;
  if (i < NNODES) dinv[i] = rsqrtf(deg[i]);
}

// ---------------------------------------------------------------------------
// Stage 1: GCN linear transform  h = x @ gcn_W   [160000x128]x[128x32]
// One wave per 16-row tile; 2 C-tiles (32 cols); K = 4 chunks of 32.
// ---------------------------------------------------------------------------
__global__ void cl_gcn_gemm(const float* __restrict__ x,
                            const half_t* __restrict__ gcnP,
                            float* __restrict__ hgcn) {
  int wave = (blockIdx.x * blockDim.x + threadIdx.x) >> 5;  // 0..9999
  int lane = threadIdx.x & 31;
  if (wave >= (NNODES * TSTEPS) / 16) return;  // wave-uniform
  const float* rowp = x + (size_t)(wave * 16 + (lane & 15)) * FEAT;
  v8f c0 = {}, c1 = {};
#pragma unroll
  for (int kc = 0; kc < 4; ++kc) {
    v16h a  = load_a_f32(rowp + kc * 32, lane);
    v16h b0 = *(const v16h*)(gcnP + ((kc * 2 + 0) * 32 + lane) * 16);
    v16h b1 = *(const v16h*)(gcnP + ((kc * 2 + 1) * 32 + lane) * 16);
    c0 = wmma_f16(a, b0, c0);
    c1 = wmma_f16(a, b1, c1);
  }
  int r0 = (lane >> 4) << 3, col = lane & 15;
  float* out = hgcn + (size_t)wave * 16 * HGCN;
#pragma unroll
  for (int v = 0; v < 8; ++v) {
    out[(r0 + v) * HGCN + col]      = c0[v];
    out[(r0 + v) * HGCN + 16 + col] = c1[v];
  }
}

// ---------------------------------------------------------------------------
// Stage 2: edge scatter  agg[col] += h[row] * dinv[row]*dinv[col]
// One thread per (edge, t); inner 32-float atomic adds (L2-resident).
// ---------------------------------------------------------------------------
__global__ void cl_message(const int* __restrict__ ei,
                           const float* __restrict__ dinv,
                           const float* __restrict__ hgcn,
                           float* __restrict__ agg) {
  int tid = blockIdx.x * blockDim.x + threadIdx.x;
  if (tid >= NEDGES * TSTEPS) return;
  int e = tid >> 3, t = tid & 7;
  int r = ei[e], c = ei[NEDGES + e];
  float norm = dinv[r] * dinv[c];
  const float* src = hgcn + ((size_t)r * TSTEPS + t) * HGCN;
  float* dst = agg + ((size_t)c * TSTEPS + t) * HGCN;
#pragma unroll
  for (int j = 0; j < HGCN; ++j) atomicAdd(dst + j, src[j] * norm);
}

// Self-loop + bias + ReLU, emit f16 for the conv GEMM A-operand.
__global__ void cl_gcn_finish(const float* __restrict__ hgcn,
                              const float* __restrict__ agg,
                              const float* __restrict__ dinv,
                              const float* __restrict__ gcn_b,
                              half_t* __restrict__ g) {
  int idx = blockIdx.x * blockDim.x + threadIdx.x;
  if (idx >= NNODES * TSTEPS * HGCN) return;
  int j = idx & 31;
  int node = idx >> 8;  // / (T*32)
  float di = dinv[node];
  float v = agg[idx] + hgcn[idx] * di * di + gcn_b[j];
  g[idx] = (half_t)fmaxf(v, 0.0f);
}

// ---------------------------------------------------------------------------
// Stage 3: Conv1d as GEMM. For (node n, out-time tau) the K=64 input vector is
// the 64 contiguous halves g[n, tau.. tau+1, :]. One wave = 16 nodes x 64 cols.
// ---------------------------------------------------------------------------
__global__ void cl_conv(const half_t* __restrict__ g,
                        const half_t* __restrict__ WcP,
                        const float* __restrict__ conv_b,
                        half_t* __restrict__ seq) {
  int wave = (blockIdx.x * blockDim.x + threadIdx.x) >> 5;
  int lane = threadIdx.x & 31;
  if (wave >= (NNODES / 16) * (TSTEPS - 1)) return;  // wave-uniform
  int tau = wave % (TSTEPS - 1);
  int n0  = (wave / (TSTEPS - 1)) * 16;
  const half_t* rowp =
      g + ((size_t)(n0 + (lane & 15)) * TSTEPS + tau) * HGCN;
  v8f c[4] = {};
#pragma unroll
  for (int kc = 0; kc < 2; ++kc) {
    v16h a = load_a_f16(rowp + kc * 32, lane);
#pragma unroll
    for (int ct = 0; ct < 4; ++ct) {
      v16h b = *(const v16h*)(WcP + ((kc * 4 + ct) * 32 + lane) * 16);
      c[ct] = wmma_f16(a, b, c[ct]);
    }
  }
  int r0 = (lane >> 4) << 3, col = lane & 15;
#pragma unroll
  for (int ct = 0; ct < 4; ++ct) {
    float bias = conv_b[ct * 16 + col];
#pragma unroll
    for (int v = 0; v < 8; ++v) {
      float y = fmaxf(c[ct][v] + bias, 0.0f);
      seq[((size_t)tau * NNODES + n0 + r0 + v) * HCONV + ct * 16 + col] =
          (half_t)y;
    }
  }
}

// ---------------------------------------------------------------------------
// Stage 4: one LSTM step. Block = 16 nodes, 8 waves. Wave w computes z columns
// [w*64, w*64+64): 4 C-tiles, K = 64 (xt) + 128 (h) = 6 chunks. z staged in
// LDS, then gate nonlinearities update h (f16) and c (f32).
// ---------------------------------------------------------------------------
__global__ void cl_lstm_step(const half_t* __restrict__ seq,
                             const half_t* __restrict__ WzP,
                             const float* __restrict__ bz,
                             half_t* __restrict__ hstate,
                             float* __restrict__ cstate, int tau) {
  __shared__ float zbuf[16][4 * HLSTM];
  int lane = threadIdx.x & 31;
  int w = threadIdx.x >> 5;  // 0..7
  int n0 = blockIdx.x * 16;
  const half_t* xrow =
      seq + ((size_t)tau * NNODES + n0 + (lane & 15)) * HCONV;
  const half_t* hrow = hstate + (size_t)(n0 + (lane & 15)) * HLSTM;
  v8f c[4] = {};
#pragma unroll
  for (int kc = 0; kc < 6; ++kc) {
    v16h a = (kc < 2) ? load_a_f16(xrow + kc * 32, lane)
                      : load_a_f16(hrow + (kc - 2) * 32, lane);
#pragma unroll
    for (int q = 0; q < 4; ++q) {
      int ct = w * 4 + q;
      v16h b = *(const v16h*)(WzP + ((size_t)(kc * 32 + ct) * 32 + lane) * 16);
      c[q] = wmma_f16(a, b, c[q]);
    }
  }
  int r0 = (lane >> 4) << 3, col = lane & 15;
#pragma unroll
  for (int q = 0; q < 4; ++q)
#pragma unroll
    for (int v = 0; v < 8; ++v)
      zbuf[r0 + v][(w * 4 + q) * 16 + col] = c[q][v];
  __syncthreads();
  // 16 nodes x 128 hidden = 2048 gate groups; 256 threads x 8 iterations.
#pragma unroll
  for (int it = 0; it < 8; ++it) {
    int idx = it * 256 + threadIdx.x;
    int n = idx >> 7, j = idx & 127;
    float zi = zbuf[n][j]       + bz[j];
    float zf = zbuf[n][128 + j] + bz[128 + j];
    float zg = zbuf[n][256 + j] + bz[256 + j];
    float zo = zbuf[n][384 + j] + bz[384 + j];
    size_t gi = (size_t)(n0 + n) * HLSTM + j;
    float cn = sigf(zf) * cstate[gi] + sigf(zi) * tanhf(zg);
    cstate[gi] = cn;
    hstate[gi] = (half_t)(sigf(zo) * tanhf(cn));
  }
}

// ---------------------------------------------------------------------------
// Stage 5: head.  out = sigmoid(relu(hT@fc1.T + b1)@fc2.T + b2)
// One wave per 16-node tile; FC1 result staged as f16 in per-wave LDS, then
// FC2 (K=64, 8 col tiles) straight to global fp32 output.
// ---------------------------------------------------------------------------
__global__ void cl_head(const half_t* __restrict__ hstate,
                        const half_t* __restrict__ fc1P,
                        const float* __restrict__ fc1_b,
                        const half_t* __restrict__ fc2P,
                        const float* __restrict__ fc2_b,
                        float* __restrict__ out) {
  __shared__ __align__(16) half_t ubuf[8][16 * 64];
  int lane = threadIdx.x & 31;
  int w = threadIdx.x >> 5;
  int tile = blockIdx.x * 8 + w;
  if (tile >= NNODES / 16) return;  // wave-uniform
  int n0 = tile * 16;
  const half_t* hrow = hstate + (size_t)(n0 + (lane & 15)) * HLSTM;
  int r0 = (lane >> 4) << 3, col = lane & 15;

  v8f c1[4] = {};
#pragma unroll
  for (int kc = 0; kc < 4; ++kc) {
    v16h a = load_a_f16(hrow + kc * 32, lane);
#pragma unroll
    for (int ct = 0; ct < 4; ++ct) {
      v16h b = *(const v16h*)(fc1P + ((kc * 4 + ct) * 32 + lane) * 16);
      c1[ct] = wmma_f16(a, b, c1[ct]);
    }
  }
  half_t* u = ubuf[w];
#pragma unroll
  for (int ct = 0; ct < 4; ++ct) {
    float bias = fc1_b[ct * 16 + col];
#pragma unroll
    for (int v = 0; v < 8; ++v)
      u[(r0 + v) * 64 + ct * 16 + col] = (half_t)fmaxf(c1[ct][v] + bias, 0.0f);
  }
  // per-wave LDS region: same-wave DS ops are in-order; no barrier needed.
  v8f c2[8] = {};
#pragma unroll
  for (int kc = 0; kc < 2; ++kc) {
    v16h a = load_a_f16(u + (lane & 15) * 64 + kc * 32, lane);
#pragma unroll
    for (int ct = 0; ct < 8; ++ct) {
      v16h b = *(const v16h*)(fc2P + ((kc * 8 + ct) * 32 + lane) * 16);
      c2[ct] = wmma_f16(a, b, c2[ct]);
    }
  }
#pragma unroll
  for (int ct = 0; ct < 8; ++ct) {
    float bias = fc2_b[ct * 16 + col];
#pragma unroll
    for (int v = 0; v < 8; ++v)
      out[(size_t)(n0 + r0 + v) * FEAT + ct * 16 + col] =
          sigf(c2[ct][v] + bias);
  }
}

// ---------------------------------------------------------------------------
// Host launcher
// ---------------------------------------------------------------------------
static inline size_t alignup256(size_t x) { return (x + 255) & ~(size_t)255; }

extern "C" void kernel_launch(void* const* d_in, const int* in_sizes, int n_in,
                              void* d_out, int out_size, void* d_ws,
                              size_t ws_size, hipStream_t stream) {
  const float* x      = (const float*)d_in[0];
  const int*   ei     = (const int*)d_in[1];
  const float* gcn_W  = (const float*)d_in[2];
  const float* gcn_b  = (const float*)d_in[3];
  const float* conv_W = (const float*)d_in[4];
  const float* conv_b = (const float*)d_in[5];
  const float* W_ih   = (const float*)d_in[6];
  const float* W_hh   = (const float*)d_in[7];
  const float* b_ih   = (const float*)d_in[8];
  const float* b_hh   = (const float*)d_in[9];
  const float* fc1_W  = (const float*)d_in[10];
  const float* fc1_b  = (const float*)d_in[11];
  const float* fc2_W  = (const float*)d_in[12];
  const float* fc2_b  = (const float*)d_in[13];
  float* out = (float*)d_out;

  // Workspace carve-up
  char* w = (char*)d_ws;
  size_t off = 0;
  auto carve = [&](size_t bytes) -> void* {
    void* p = w + off;
    off = alignup256(off + bytes);
    return p;
  };
  const size_t NT32 = (size_t)NNODES * TSTEPS * HGCN;  // 5.12M
  float*  deg    = (float*)carve((size_t)NNODES * 4);
  float*  dinv   = (float*)carve((size_t)NNODES * 4);
  float*  hgcn   = (float*)carve(NT32 * 4);
  float*  agg    = (float*)carve(NT32 * 4);
  half_t* gbuf   = (half_t*)carve(NT32 * 2);
  half_t* seq    = (half_t*)carve((size_t)(TSTEPS - 1) * NNODES * HCONV * 2);
  half_t* hstate = (half_t*)carve((size_t)NNODES * HLSTM * 2);
  float*  cstate = (float*)carve((size_t)NNODES * HLSTM * 4);
  half_t* gcnP   = (half_t*)carve(4096 * 2);
  half_t* WcP    = (half_t*)carve(4096 * 2);
  half_t* WzP    = (half_t*)carve(6 * 32 * 512 * 2);  // K=192 -> 6 chunks
  half_t* fc1P   = (half_t*)carve(8192 * 2);
  half_t* fc2P   = (half_t*)carve(8192 * 2);
  float*  bz     = (float*)carve(512 * 4);
  (void)ws_size; (void)n_in; (void)in_sizes; (void)out_size;

  auto blk = [](size_t n) { return (unsigned)((n + 255) / 256); };

  // ---- init / prep ----
  cl_fill_f32<<<blk(NNODES), 256, 0, stream>>>(deg, 1.0f, NNODES);  // self-loops
  cl_fill_f32<<<blk(NT32), 256, 0, stream>>>(agg, 0.0f, (int)NT32);
  cl_fill_f32<<<blk((size_t)NNODES * HLSTM), 256, 0, stream>>>(
      cstate, 0.0f, NNODES * HLSTM);
  cl_fill_f16<<<blk((size_t)NNODES * HLSTM), 256, 0, stream>>>(
      hstate, NNODES * HLSTM);

  cl_pack_b<<<blk(4096), 256, 0, stream>>>(gcn_W, gcnP, 128, 32, 1);
  cl_pack_conv<<<blk(4096), 256, 0, stream>>>(conv_W, WcP);
  cl_pack_b<<<blk(2 * 32 * 512), 256, 0, stream>>>(W_ih, WzP, 64, 512, 0);
  cl_pack_b<<<blk(4 * 32 * 512), 256, 0, stream>>>(W_hh, WzP + 2 * 32 * 512,
                                                   128, 512, 0);
  cl_pack_b<<<blk(8192), 256, 0, stream>>>(fc1_W, fc1P, 128, 64, 0);
  cl_pack_b<<<blk(8192), 256, 0, stream>>>(fc2_W, fc2P, 64, 128, 0);
  cl_bias_sum<<<2, 256, 0, stream>>>(b_ih, b_hh, bz);

  cl_degree<<<blk(NEDGES), 256, 0, stream>>>(ei, deg);
  cl_dinv<<<blk(NNODES), 256, 0, stream>>>(deg, dinv);

  // ---- GCN transform (WMMA) ----
  cl_gcn_gemm<<<(NNODES * TSTEPS / 16) / 4, 128, 0, stream>>>(x, gcnP, hgcn);

  // ---- edge scatter + finish ----
  cl_message<<<blk((size_t)NEDGES * TSTEPS), 256, 0, stream>>>(ei, dinv, hgcn,
                                                               agg);
  cl_gcn_finish<<<blk(NT32), 256, 0, stream>>>(hgcn, agg, dinv, gcn_b, gbuf);

  // ---- Conv1d as GEMM (WMMA) ----
  {
    size_t waves = (size_t)(NNODES / 16) * (TSTEPS - 1);  // 8750
    cl_conv<<<(unsigned)((waves * 32 + 255) / 256), 256, 0, stream>>>(
        gbuf, WcP, conv_b, seq);
  }

  // ---- LSTM: 7 sequential steps (WMMA) ----
  for (int tau = 0; tau < TSTEPS - 1; ++tau)
    cl_lstm_step<<<NNODES / 16, 256, 0, stream>>>(seq, WzP, bz, hstate, cstate,
                                                  tau);

  // ---- Head (WMMA) ----
  cl_head<<<(NNODES / 16 + 7) / 8, 256, 0, stream>>>(hstate, fc1P, fc1_b, fc2P,
                                                     fc2_b, out);
}
